// Model_39676907884251
// MI455X (gfx1250) — matrix operand order repaired and checked by reading, not compiled, not run on hardware
//
#include <hip/hip_runtime.h>

// ---- CDNA5 WMMA types ----
typedef __attribute__((ext_vector_type(16))) __bf16 v16bf;
typedef __attribute__((ext_vector_type(8)))  float  v8f;

typedef unsigned short ushort_t;
typedef unsigned int   uint_t;

#define BATCH  2
#define SEQ    2048
#define NH     16
#define DH     128
#define DMODEL (NH * DH)

#define QROWS_PER_WG 128   // 8 waves x 16 q-rows
#define WAVES        8
#define KT           32    // keys per kv tile (WMMA K=32 for PV gemm)
#define NTILES       (SEQ / KT)

union FragAB { v16bf v; ushort_t s[16]; uint4 q[2]; };
union FragC  { v8f   v; float    f[8]; };
union Pack8  { ushort_t s[16]; uint4 q[2]; };

// fp32 -> bf16, round-to-nearest-even
__device__ __forceinline__ ushort_t f2bf(float f) {
    uint_t u = __builtin_bit_cast(uint_t, f);
    u += 0x7FFFu + ((u >> 16) & 1u);
    return (ushort_t)(u >> 16);
}

// ---------- 16-lane butterfly reductions (DPP: pure VALU, co-executes with WMMA) ----------
#if __has_builtin(__builtin_amdgcn_update_dpp)
#define DPP_STEP_F32(x, ctrl, op)                                              \
    do {                                                                       \
        int _t = __builtin_amdgcn_update_dpp(0, __builtin_bit_cast(int, (x)),  \
                                             (ctrl), 0xF, 0xF, true);          \
        (x) = op((x), __builtin_bit_cast(float, _t));                          \
    } while (0)
__device__ __forceinline__ float fadd_(float a, float b) { return a + b; }
__device__ __forceinline__ float red_max16(float x) {
    DPP_STEP_F32(x, 0xB1,  fmaxf);  // quad_perm [1,0,3,2]  : xor 1
    DPP_STEP_F32(x, 0x4E,  fmaxf);  // quad_perm [2,3,0,1]  : xor 2
    DPP_STEP_F32(x, 0x141, fmaxf);  // row_half_mirror      : reduce 8
    DPP_STEP_F32(x, 0x140, fmaxf);  // row_mirror           : reduce 16
    return x;
}
__device__ __forceinline__ float red_sum16(float x) {
    DPP_STEP_F32(x, 0xB1,  fadd_);
    DPP_STEP_F32(x, 0x4E,  fadd_);
    DPP_STEP_F32(x, 0x141, fadd_);
    DPP_STEP_F32(x, 0x140, fadd_);
    return x;
}
#else
__device__ __forceinline__ float red_max16(float v) {
#pragma unroll
    for (int m = 1; m <= 8; m <<= 1) v = fmaxf(v, __shfl_xor(v, m, 32));
    return v;
}
__device__ __forceinline__ float red_sum16(float v) {
#pragma unroll
    for (int m = 1; m <= 8; m <<= 1) v += __shfl_xor(v, m, 32);
    return v;
}
#endif

// ---------- async global -> LDS copy (CDNA5 ASYNCcnt path) ----------
// Signature (confirmed via hipcc diagnostic): (global v4i*, shared v4i*, i32 off, i32 cpol)
#if __has_builtin(__builtin_amdgcn_global_load_async_to_lds_b128) && \
    __has_builtin(__builtin_amdgcn_s_wait_asynccnt)
#define USE_ASYNC_LDS 1
typedef int v4i_t __attribute__((__vector_size__(16)));
typedef __attribute__((address_space(1))) v4i_t* gvec_ptr_t;
typedef __attribute__((address_space(3))) v4i_t* lvec_ptr_t;
__device__ __forceinline__ void async_cp16(const void* g, void* l) {
    gvec_ptr_t gp = (gvec_ptr_t)(unsigned long long)g;
    lvec_ptr_t lp = (lvec_ptr_t)(unsigned int)(unsigned long long)l;
    __builtin_amdgcn_global_load_async_to_lds_b128(gp, lp, 0, 0);
}
__device__ __forceinline__ void async_wait0() {
    __builtin_amdgcn_s_wait_asynccnt(0);
}
#else
#define USE_ASYNC_LDS 0
__device__ __forceinline__ void async_cp16(const void* g, void* l) {
    *(uint4*)l = *(const uint4*)g;   // sync fallback
}
__device__ __forceinline__ void async_wait0() {}
#endif

// =====================================================================
// Pre-pass: K -> bf16 [b,h,s,dh] ; V -> bf16 transposed [b,h,dh,s]
// Memory-bound one-shot (~100MB ~ 4.3us @ 23.3 TB/s); removes all
// fp32->bf16 VALU work from the flash inner loop and halves staged bytes.
// =====================================================================
__global__ __launch_bounds__(256)
void cvt_kv(const float* __restrict__ Kg, const float* __restrict__ Vg,
            ushort_t* __restrict__ Kbf, ushort_t* __restrict__ VTbf)
{
    __shared__ ushort_t tv[DH * KT];   // transposed V tile [dh][key], 8 KB

    const int tid = threadIdx.x;
    const int b   = blockIdx.z;
    const int h   = blockIdx.y;
    const int kt0 = blockIdx.x * KT;

    const int r32 = tid >> 3;   // key row 0..31
    const int cb  = tid & 7;    // 16-float col block

    const size_t goff = ((size_t)(b * SEQ + kt0 + r32)) * DMODEL + h * DH + cb * 16;
    const float4* kp = (const float4*)(Kg + goff);
    const float4* vp = (const float4*)(Vg + goff);

    Pack8 krow;
#pragma unroll
    for (int j4 = 0; j4 < 4; ++j4) {
        float4 kf = kp[j4];
        float4 vf = vp[j4];
        krow.s[j4 * 4 + 0] = f2bf(kf.x);
        krow.s[j4 * 4 + 1] = f2bf(kf.y);
        krow.s[j4 * 4 + 2] = f2bf(kf.z);
        krow.s[j4 * 4 + 3] = f2bf(kf.w);
        const int col = cb * 16 + j4 * 4;
        tv[(col + 0) * KT + r32] = f2bf(vf.x);
        tv[(col + 1) * KT + r32] = f2bf(vf.y);
        tv[(col + 2) * KT + r32] = f2bf(vf.z);
        tv[(col + 3) * KT + r32] = f2bf(vf.w);
    }
    {
        uint4* kd = (uint4*)(Kbf + ((size_t)((b * NH + h) * SEQ) + kt0 + r32) * DH + cb * 16);
        kd[0] = krow.q[0];
        kd[1] = krow.q[1];
    }
    __syncthreads();
    {
        const int dh   = tid >> 1;
        const int half = tid & 1;
        const uint4* src = (const uint4*)&tv[dh * KT + half * 16];
        uint4* dst = (uint4*)(VTbf + ((size_t)((b * NH + h) * DH) + dh) * SEQ + kt0 + half * 16);
        dst[0] = src[0];
        dst[1] = src[1];
    }
}

// =====================================================================
// Flash attention forward, bf16 WMMA, fp32 accumulate.
// PRE=true : bf16 K/V^T in workspace; double-buffered async global->LDS
//            pipeline overlaps the next tile's DMA with this tile's WMMAs.
// PRE=false: stage+convert from fp32, single buffer (ws too small).
// =====================================================================
template <bool PRE>
__global__ __launch_bounds__(WAVES * 32)
void fa_fwd(const float* __restrict__ Qg,
            const float* __restrict__ Kg, const float* __restrict__ Vg,
            const ushort_t* __restrict__ Kbf, const ushort_t* __restrict__ VTbf,
            float* __restrict__ Og)
{
    __shared__ ushort_t lK [2][KT * DH];      // K tiles  bf16 [key][dh]  16 KB
    __shared__ ushort_t lVT[2][DH * KT];      // V tiles  bf16 [dh][key]  16 KB
    __shared__ ushort_t lP [WAVES * 16 * KT]; // per-wave P scratch        8 KB

    const int tid  = threadIdx.x;
    const int wave = tid >> 5;
    const int lane = tid & 31;
    const int n    = lane & 15;
    const int hi   = lane >> 4;

    const int b  = blockIdx.z;
    const int h  = blockIdx.y;
    const int qg = blockIdx.x * QROWS_PER_WG + wave * 16;

    const float scale = 0.08838834764831845f; // 1/sqrt(128)

    // per-thread staging source pointers (PRE path)
    const int dhv  = tid >> 1;
    const int half = tid & 1;
    const ushort_t* ksrc0 = PRE ? (Kbf + ((size_t)((b * NH + h) * SEQ)) * DH + tid * 16) : nullptr;
    const ushort_t* vsrc0 = PRE ? (VTbf + ((size_t)((b * NH + h) * DH) + dhv) * SEQ + half * 16)
                                : nullptr;

    // Q A-fragments (contiguous k order used by this code: lane -> M = L%16, K = hi*16 + {0..15})
    FragAB aq[4];
    {
        const float* qrow = Qg + ((size_t)(b * SEQ + qg + n)) * DMODEL + h * DH;
        const int b0 = hi * 16;
#pragma unroll
        for (int c = 0; c < 4; ++c) {
#pragma unroll
            for (int j = 0; j < 8; ++j) {
                aq[c].s[j]     = f2bf(qrow[c * 32 + b0 + j]      * scale);
                aq[c].s[8 + j] = f2bf(qrow[c * 32 + b0 + 8 + j] * scale);
            }
        }
    }

    FragC acc[8];
#pragma unroll
    for (int t = 0; t < 8; ++t)
#pragma unroll
        for (int r = 0; r < 8; ++r) acc[t].f[r] = 0.0f;

    float mrow[8], lrow[8];
#pragma unroll
    for (int r = 0; r < 8; ++r) { mrow[r] = -3.0e38f; lrow[r] = 0.0f; }

    const int r32 = tid >> 3;
    const int cb  = tid & 7;

    // -------- pipeline prologue: stage tile 0 into buffer 0 --------
    if (PRE) {
        async_cp16(ksrc0,     &lK [0][tid * 16]);
        async_cp16(ksrc0 + 8, &lK [0][tid * 16 + 8]);
        async_cp16(vsrc0,     &lVT[0][dhv * KT + half * 16]);
        async_cp16(vsrc0 + 8, &lVT[0][dhv * KT + half * 16 + 8]);
        async_wait0();
        __syncthreads();
    }

    for (int kt = 0; kt < NTILES; ++kt) {
        const int cur = PRE ? (kt & 1) : 0;

        if (PRE) {
            // ---- issue async DMA of tile kt+1 into the other buffer ----
            if (kt + 1 < NTILES) {
                const int nxt = cur ^ 1;
                const ushort_t* ks = ksrc0 + (size_t)(kt + 1) * KT * DH;
                const ushort_t* vs = vsrc0 + (size_t)(kt + 1) * KT;
                async_cp16(ks,     &lK [nxt][tid * 16]);
                async_cp16(ks + 8, &lK [nxt][tid * 16 + 8]);
                async_cp16(vs,     &lVT[nxt][dhv * KT + half * 16]);
                async_cp16(vs + 8, &lVT[nxt][dhv * KT + half * 16 + 8]);
            }
        } else {
            // ---- fallback: stage + convert fp32 -> bf16 into buffer 0 ----
            const size_t goff = ((size_t)(b * SEQ + kt * KT + r32)) * DMODEL + h * DH + cb * 16;
            const float4* kp = (const float4*)(Kg + goff);
            const float4* vp = (const float4*)(Vg + goff);
#pragma unroll
            for (int j4 = 0; j4 < 4; ++j4) {
                float4 kf = kp[j4];
                float4 vf = vp[j4];
                const int col = cb * 16 + j4 * 4;
                lK[0][r32 * DH + col + 0] = f2bf(kf.x);
                lK[0][r32 * DH + col + 1] = f2bf(kf.y);
                lK[0][r32 * DH + col + 2] = f2bf(kf.z);
                lK[0][r32 * DH + col + 3] = f2bf(kf.w);
                lVT[0][(col + 0) * KT + r32] = f2bf(vf.x);
                lVT[0][(col + 1) * KT + r32] = f2bf(vf.y);
                lVT[0][(col + 2) * KT + r32] = f2bf(vf.z);
                lVT[0][(col + 3) * KT + r32] = f2bf(vf.w);
            }
            if (kt + 1 < NTILES) {
                __builtin_prefetch(Kg + goff + (size_t)KT * DMODEL, 0, 0);
                __builtin_prefetch(Vg + goff + (size_t)KT * DMODEL, 0, 0);
            }
            __syncthreads();
        }

        // ---------------- S = (Q*scale) K^T ----------------
        FragC s0, s1;
#pragma unroll
        for (int r = 0; r < 8; ++r) { s0.f[r] = 0.0f; s1.f[r] = 0.0f; }
#pragma unroll
        for (int c = 0; c < 4; ++c) {
            FragAB bk0, bk1;
            const uint4* p0 = (const uint4*)&lK[cur][(n)      * DH + c * 32 + hi * 16];
            const uint4* p1 = (const uint4*)&lK[cur][(16 + n) * DH + c * 32 + hi * 16];
            bk0.q[0] = p0[0]; bk0.q[1] = p0[1];
            bk1.q[0] = p1[0]; bk1.q[1] = p1[1];
            s0.v = __builtin_amdgcn_wmma_f32_16x16x32_bf16(false, aq[c].v, false, bk0.v,
                                                           (short)0, s0.v, false, false);
            s1.v = __builtin_amdgcn_wmma_f32_16x16x32_bf16(false, aq[c].v, false, bk1.v,
                                                           (short)0, s1.v, false, false);
        }

        // ---------------- online softmax (row = r + 8*hi, col = n) ----------------
#pragma unroll
        for (int r = 0; r < 8; ++r) {
            float mx   = red_max16(fmaxf(s0.f[r], s1.f[r]));
            float mnew = fmaxf(mrow[r], mx);
            float p0   = __expf(s0.f[r] - mnew);
            float p1   = __expf(s1.f[r] - mnew);
            float rs   = red_sum16(p0 + p1);
            float alpha = __expf(mrow[r] - mnew);
            lrow[r] = lrow[r] * alpha + rs;
            mrow[r] = mnew;
#pragma unroll
            for (int t = 0; t < 8; ++t) acc[t].f[r] *= alpha;
            const int row = r + 8 * hi;
            lP[wave * (16 * KT) + row * KT + n]      = f2bf(p0);
            lP[wave * (16 * KT) + row * KT + 16 + n] = f2bf(p1);
        }

        // ---------------- O += P V ----------------
        FragAB ap;
        {
            const uint4* pp0 = (const uint4*)&lP[wave * (16 * KT) + n * KT + hi * 16];
            const uint4* pp1 = (const uint4*)&lP[wave * (16 * KT) + n * KT + hi * 16 + 8];
            ap.q[0] = pp0[0];
            ap.q[1] = pp1[0];
        }
#pragma unroll
        for (int t = 0; t < 8; ++t) {
            FragAB bv;
            const uint4* pv = (const uint4*)&lVT[cur][(t * 16 + n) * KT + hi * 16];
            bv.q[0] = pv[0]; bv.q[1] = pv[1];
            acc[t].v = __builtin_amdgcn_wmma_f32_16x16x32_bf16(false, ap.v, false, bv.v,
                                                               (short)0, acc[t].v, false, false);
        }

        // ---- pipeline epilogue of this iteration ----
        if (PRE) async_wait0();   // my async writes to buf^1 have landed
        __syncthreads();          // everyone done reading buf[cur] / writes visible
    }

    // ---------------- epilogue ----------------
    float inv[8];
#pragma unroll
    for (int r = 0; r < 8; ++r) inv[r] = 1.0f / lrow[r];
#pragma unroll
    for (int t = 0; t < 8; ++t) {
#pragma unroll
        for (int r = 0; r < 8; ++r) {
            const size_t off = ((size_t)(b * SEQ + qg + r + 8 * hi)) * DMODEL
                             + h * DH + t * 16 + n;
            Og[off] = acc[t].f[r] * inv[r];
        }
    }
}

extern "C" void kernel_launch(void* const* d_in, const int* in_sizes, int n_in,
                              void* d_out, int out_size, void* d_ws, size_t ws_size,
                              hipStream_t stream) {
    (void)in_sizes; (void)n_in; (void)out_size;
    const float* Q = (const float*)d_in[0];
    const float* K = (const float*)d_in[1];
    const float* V = (const float*)d_in[2];
    // d_in[3] = dropout_p == 0 -> identity (ignored)
    float* O = (float*)d_out;

    dim3 grid(SEQ / QROWS_PER_WG, NH, BATCH);  // (16, 16, 2)
    dim3 block(WAVES * 32);                    // 256 threads = 8 wave32

    const size_t kv_elems = (size_t)BATCH * NH * SEQ * DH;
    const size_t need     = 2 * kv_elems * sizeof(ushort_t);  // 32 MiB

    if (ws_size >= need) {
        ushort_t* Kbf  = (ushort_t*)d_ws;
        ushort_t* VTbf = Kbf + kv_elems;
        cvt_kv<<<dim3(SEQ / KT, NH, BATCH), 256, 0, stream>>>(K, V, Kbf, VTbf);
        fa_fwd<true><<<grid, block, 0, stream>>>(Q, nullptr, nullptr, Kbf, VTbf, O);
    } else {
        fa_fwd<false><<<grid, block, 0, stream>>>(Q, K, V, nullptr, nullptr, O);
    }
}
